// GAT_Encoder_55035710931804
// MI455X (gfx1250) — matrix operand order repaired and checked by reading, not compiled, not run on hardware
//
#include <hip/hip_runtime.h>
#include <hip/hip_bf16.h>
#include <stdint.h>

typedef __attribute__((ext_vector_type(16))) __bf16 v16bf;
typedef __attribute__((ext_vector_type(8)))  float  v8f;

__device__ __forceinline__ unsigned short f2bf(float f) {
  unsigned int u = __float_as_uint(f);
  unsigned int r = u + 0x7fffu + ((u >> 16) & 1u);   // round-to-nearest-even
  return (unsigned short)(r >> 16);
}
__device__ __forceinline__ int fkey(float f) {      // monotone float->int key
  int b = __float_as_int(f);
  return b >= 0 ? b : (b ^ 0x7fffffff);
}
__device__ __forceinline__ float keyf(int k) {
  return __int_as_float(k >= 0 ? k : (k ^ 0x7fffffff));
}

// ---------- fp32 -> bf16 convert ----------
__global__ void k_f2bf(const float* __restrict__ in, unsigned short* __restrict__ out, int n) {
  int i = blockIdx.x * blockDim.x + threadIdx.x;
  if (i < n) out[i] = f2bf(in[i]);
}

// ---------- stack [Wl;Wr] -> bf16 weight matrix [512 x K] row-major ----------
__global__ void k_wcat(const float* __restrict__ Wl, const float* __restrict__ Wr,
                       unsigned short* __restrict__ wbf, int K) {
  int i = blockIdx.x * blockDim.x + threadIdx.x;
  int n = 512 * K;
  if (i >= n) return;
  int r = i / K, k = i - r * K;
  float v = (r < 256) ? Wl[r * K + k] : Wr[(r - 256) * K + k];
  wbf[i] = f2bf(v);
}

// ---------- bf16 WMMA GEMM: C[N x 512] = A[N x K] * W[512 x K]^T + bias ----------
union Frag { uint4 u[2]; v16bf v; };

__global__ void k_gemm(const unsigned short* __restrict__ A,
                       const unsigned short* __restrict__ W,
                       const float* __restrict__ bl, const float* __restrict__ br,
                       float* __restrict__ C, int nrows, int K) {
  int lane = threadIdx.x & 31;
  int wave = threadIdx.x >> 5;
  int hf   = lane >> 4;           // 0 or 1
  int l16  = lane & 15;
  int row0 = blockIdx.x * 16;
  int colG = (blockIdx.y * 4 + wave) * 16 + l16;   // 0..511
  int row  = row0 + l16;
  if (row >= nrows) row = nrows - 1;

  const unsigned short* arow = A + (size_t)row  * K;
  const unsigned short* wrow = W + (size_t)colG * K;
  v8f acc = {};
  for (int k0 = 0; k0 < K; k0 += 32) {
    Frag a, b;
    // A 16x32 bf16: elems 0..7 -> K = hf*8 + j ; elems 8..15 -> K = 16 + hf*8 + j
    a.u[0] = *(const uint4*)(arow + k0 + hf * 8);
    a.u[1] = *(const uint4*)(arow + k0 + 16 + hf * 8);
    // B 32x16 bf16: lane col = lane&15, elems 0..7 -> K = hf*8 + j ; elems 8..15 -> K = 16 + hf*8 + j
    b.u[0] = *(const uint4*)(wrow + k0 + hf * 8);
    b.u[1] = *(const uint4*)(wrow + k0 + 16 + hf * 8);
    acc = __builtin_amdgcn_wmma_f32_16x16x32_bf16(false, a.v, false, b.v,
                                                  (short)0, acc, false, false);
  }
  float bias = (colG < 256) ? bl[colG] : br[colG - 256];
  int mbase = row0 + 8 * hf;      // D: vgpr v -> M = v + 8*hf
#pragma unroll
  for (int v = 0; v < 8; ++v) {
    int m = mbase + v;
    if (m < nrows) C[(size_t)m * 512 + colG] = acc[v] + bias;
  }
}

// ---------- per-layer init ----------
__global__ void k_init(int* __restrict__ mkey, float* __restrict__ sden,
                       float* __restrict__ acc, int N) {
  int i = blockIdx.x * blockDim.x + threadIdx.x;
  if (i >= N * 256) return;
  acc[i] = 0.f;
  if (i < N * 4) { mkey[i] = (int)0x80000000; sden[i] = 0.f; }
}

// ---------- pass 1: edge scores + segment max (one wave per edge) ----------
__global__ void k_edge_score(const int* __restrict__ ei, int E, int ET,
                             const float* __restrict__ xlr,
                             const float* __restrict__ att,
                             float* __restrict__ score, int* __restrict__ mkey) {
  int lane = threadIdx.x & 31;
  int e = blockIdx.x * 8 + (threadIdx.x >> 5);
  if (e >= ET) return;
  int src, dst;
  if (e < E) { src = ei[e]; dst = ei[E + e]; } else { src = e - E; dst = src; }
  int cb = lane * 8;              // channels [8*lane, 8*lane+8), head = lane>>3
  const float4* pl = (const float4*)(xlr + (size_t)src * 512 + cb);
  const float4* pr = (const float4*)(xlr + (size_t)dst * 512 + 256 + cb);
  const float4* pa = (const float4*)(att + cb);
  float part = 0.f;
#pragma unroll
  for (int q = 0; q < 2; ++q) {
    float4 vl = pl[q], vr = pr[q], va = pa[q];
    float t;
    t = vl.x + vr.x; t = t > 0.f ? t : 0.2f * t; part += va.x * t;
    t = vl.y + vr.y; t = t > 0.f ? t : 0.2f * t; part += va.y * t;
    t = vl.z + vr.z; t = t > 0.f ? t : 0.2f * t; part += va.z * t;
    t = vl.w + vr.w; t = t > 0.f ? t : 0.2f * t; part += va.w * t;
  }
  part += __shfl_xor(part, 1, 32);
  part += __shfl_xor(part, 2, 32);
  part += __shfl_xor(part, 4, 32);  // now all 8 lanes of a head group hold the score
  if ((lane & 7) == 0) {
    int head = lane >> 3;
    score[(size_t)e * 4 + head] = part;
    atomicMax(&mkey[dst * 4 + head], fkey(part));
  }
}

// ---------- pass 2: p = exp(score - m), scatter p and p*xl[src] ----------
__global__ void k_edge_scatter(const int* __restrict__ ei, int E, int ET,
                               const float* __restrict__ xlr,
                               const float* __restrict__ score,
                               const int* __restrict__ mkey,
                               float* __restrict__ sden, float* __restrict__ acc) {
  int lane = threadIdx.x & 31;
  int e = blockIdx.x * 8 + (threadIdx.x >> 5);
  if (e >= ET) return;
  int src, dst;
  if (e < E) { src = ei[e]; dst = ei[E + e]; } else { src = e - E; dst = src; }
  int head = lane >> 3;
  float m = keyf(mkey[dst * 4 + head]);
  float p = __expf(score[(size_t)e * 4 + head] - m);
  if ((lane & 7) == 0) atomicAdd(&sden[dst * 4 + head], p);
  int cb = lane * 8;
  const float4* pl = (const float4*)(xlr + (size_t)src * 512 + cb);
  float* ad = acc + (size_t)dst * 256 + cb;
  float4 v0 = pl[0], v1 = pl[1];
  atomicAdd(ad + 0, p * v0.x); atomicAdd(ad + 1, p * v0.y);
  atomicAdd(ad + 2, p * v0.z); atomicAdd(ad + 3, p * v0.w);
  atomicAdd(ad + 4, p * v1.x); atomicAdd(ad + 5, p * v1.y);
  atomicAdd(ad + 6, p * v1.z); atomicAdd(ad + 7, p * v1.w);
}

// ---------- pass 3: normalize + bias (+ LeakyReLU 0.01) -> bf16 for next GEMM ----------
__global__ void k_finalize_mid(const float* __restrict__ acc, const float* __restrict__ sden,
                               const float* __restrict__ bo, unsigned short* __restrict__ hbf,
                               int N) {
  int i = blockIdx.x * blockDim.x + threadIdx.x;
  if (i >= N * 256) return;
  int n = i >> 8, ch = i & 255, head = ch >> 6;
  float v = acc[i] / sden[n * 4 + head] + bo[ch];
  v = v > 0.f ? v : 0.01f * v;
  hbf[i] = f2bf(v);
}
__global__ void k_finalize_last(float* __restrict__ acc, const float* __restrict__ sden,
                                const float* __restrict__ bo, int N) {
  int i = blockIdx.x * blockDim.x + threadIdx.x;
  if (i >= N * 256) return;
  int n = i >> 8, ch = i & 255, head = ch >> 6;
  acc[i] = acc[i] / sden[n * 4 + head] + bo[ch];
}

// ---------- global mean pool ----------
__global__ void k_pool_init(float* __restrict__ pool, float* __restrict__ cnt) {
  int i = blockIdx.x * blockDim.x + threadIdx.x;
  if (i < 32 * 256) pool[i] = 0.f;
  if (i < 32) cnt[i] = 0.f;
}
__global__ void k_pool_acc(const float* __restrict__ h, const int* __restrict__ batch,
                           float* __restrict__ pool, float* __restrict__ cnt, int N) {
  int i = blockIdx.x * blockDim.x + threadIdx.x;
  if (i >= N * 256) return;
  int n = i >> 8, ch = i & 255;
  int g = batch[n];
  atomicAdd(&pool[g * 256 + ch], h[i]);
  if (ch == 0) atomicAdd(&cnt[g], 1.f);
}
__global__ void k_pool_div(const float* __restrict__ pool, const float* __restrict__ cnt,
                           float* __restrict__ out) {
  int i = blockIdx.x * blockDim.x + threadIdx.x;
  if (i >= 32 * 256) return;
  float c = cnt[i >> 8];
  out[i] = pool[i] / (c > 1.f ? c : 1.f);
}

extern "C" void kernel_launch(void* const* d_in, const int* in_sizes, int n_in,
                              void* d_out, int out_size, void* d_ws, size_t ws_size,
                              hipStream_t stream) {
  const float* x    = (const float*)d_in[0];
  const int* ei     = (const int*)d_in[1];
  const int* batch  = (const int*)d_in[2];
  const int N  = in_sizes[0] / 128;
  const int E  = in_sizes[1] / 2;
  const int ET = E + N;

  const float* Wl[3] = {(const float*)d_in[3],  (const float*)d_in[9],  (const float*)d_in[15]};
  const float* bl[3] = {(const float*)d_in[4],  (const float*)d_in[10], (const float*)d_in[16]};
  const float* Wr[3] = {(const float*)d_in[5],  (const float*)d_in[11], (const float*)d_in[17]};
  const float* br[3] = {(const float*)d_in[6],  (const float*)d_in[12], (const float*)d_in[18]};
  const float* at[3] = {(const float*)d_in[7],  (const float*)d_in[13], (const float*)d_in[19]};
  const float* bo[3] = {(const float*)d_in[8],  (const float*)d_in[14], (const float*)d_in[20]};
  const int fin[3] = {128, 256, 256};

  char* ws = (char*)d_ws;
  size_t off = 0;
  auto alloc = [&](size_t b) -> char* {
    char* p = ws + off; off = (off + b + 255) & ~(size_t)255; return p;
  };
  unsigned short* Hbf = (unsigned short*)alloc((size_t)N * 256 * 2);   // layer input, bf16
  float*          xlr = (float*)alloc((size_t)N * 512 * 4);            // [xl | xr] fp32
  float*          acc = (float*)alloc((size_t)N * 256 * 4);            // segment sums
  float*          sco = (float*)alloc((size_t)ET * 4 * 4);             // edge scores
  int*            mky = (int*)alloc((size_t)N * 4 * 4);                // float-max keys
  float*          sdn = (float*)alloc((size_t)N * 4 * 4);              // softmax denoms
  unsigned short* wbf = (unsigned short*)alloc((size_t)512 * 256 * 2); // stacked weights bf16
  float*          pol = (float*)alloc((size_t)32 * 256 * 4);
  float*          cnt = (float*)alloc((size_t)32 * 4);
  (void)ws_size; (void)n_in; (void)out_size;

  // input features -> bf16
  { int n = N * 128; k_f2bf<<<(n + 255) / 256, 256, 0, stream>>>(x, Hbf, n); }

  for (int li = 0; li < 3; ++li) {
    int K = fin[li];
    { int n = 512 * K; k_wcat<<<(n + 255) / 256, 256, 0, stream>>>(Wl[li], Wr[li], wbf, K); }
    { dim3 g((N + 15) / 16, 8);
      k_gemm<<<g, 128, 0, stream>>>(Hbf, wbf, bl[li], br[li], xlr, N, K); }
    { int n = N * 256; k_init<<<(n + 255) / 256, 256, 0, stream>>>(mky, sdn, acc, N); }
    { int nb = (ET + 7) / 8;
      k_edge_score<<<nb, 256, 0, stream>>>(ei, E, ET, xlr, at[li], sco, mky);
      k_edge_scatter<<<nb, 256, 0, stream>>>(ei, E, ET, xlr, sco, mky, sdn, acc); }
    { int n = N * 256;
      if (li < 2) k_finalize_mid<<<(n + 255) / 256, 256, 0, stream>>>(acc, sdn, bo[li], Hbf, N);
      else        k_finalize_last<<<(n + 255) / 256, 256, 0, stream>>>(acc, sdn, bo[li], N); }
  }

  k_pool_init<<<(32 * 256 + 255) / 256, 256, 0, stream>>>(pol, cnt);
  { int n = N * 256; k_pool_acc<<<(n + 255) / 256, 256, 0, stream>>>(acc, batch, pol, cnt, N); }
  k_pool_div<<<(32 * 256 + 255) / 256, 256, 0, stream>>>(pol, cnt, (float*)d_out);
}
